// Net_72782515799010
// MI455X (gfx1250) — matrix operand order repaired and checked by reading, not compiled, not run on hardware
//
#include <hip/hip_runtime.h>
#include <hip/hip_fp16.h>

// ---------------- problem constants (from reference) ----------------
#define NN 20000
#define EE 500000
#define KK 48            // 3*8*2 spline kernels
#define R1 49            // (K+1)*c_in, layer1 c_in=1
#define R2 1568          // (K+1)*32
#define R3 3136          // (K+1)*64

typedef _Float16 v2h  __attribute__((ext_vector_type(2)));
typedef _Float16 v8h  __attribute__((ext_vector_type(8)));
typedef _Float16 v16h __attribute__((ext_vector_type(16)));
typedef float    v8f  __attribute__((ext_vector_type(8)));

// ---------------- packed f16 atomic add: explicit gfx1250 instruction ----------------
__device__ inline void pk_atomic_add(_Float16* p, float a, float b) {
  union { v2h h; unsigned u; } v;
  v.h = (v2h){(_Float16)a, (_Float16)b};
  asm volatile("global_atomic_pk_add_f16 %0, %1, off" :: "v"(p), "v"(v.u) : "memory");
}

// ---------------- degree-1 B-spline basis, fracs computed once per edge -------------
struct Basis {
  float f0, f1, f2;
  int   i0, i1, i2;
  __device__ void init(const float* __restrict__ ps, int e) {
    const float inv = 1.0f / 4.5f;
    float u0 = fminf(fmaxf(ps[e * 3 + 0] * inv, 0.f), 1.f);
    float u1 = fminf(fmaxf(ps[e * 3 + 1] * inv, 0.f), 1.f);
    float u2 = fminf(fmaxf(ps[e * 3 + 2] * inv, 0.f), 1.f);
    float p0 = u0 * 2.0f, p1 = u1 * 7.0f, p2 = u2 * 1.0f;
    float l0 = floorf(p0), l1 = floorf(p1), l2 = floorf(p2);
    f0 = p0 - l0; f1 = p1 - l1; f2 = p2 - l2;
    i0 = (int)l0; i1 = (int)l1; i2 = (int)l2;
  }
  __device__ void corner(int s, float& w, int& kid) const {
    int b0 = s & 1, b1 = (s >> 1) & 1, b2 = (s >> 2) & 1;
    int j0 = min(i0 + b0, 2);
    int j1 = min(i1 + b1, 7);
    int j2 = min(i2 + b2, 1);
    w = (b0 ? f0 : 1.f - f0) * (b1 ? f1 : 1.f - f1) * (b2 ? f2 : 1.f - f2);
    kid = j0 * 16 + j1 * 2 + j2;   // strides (16,2,1)
  }
};

// ---------------- workspace zeroing ----------------
__global__ void zero_bytes_k(uint4* p, size_t n16) {
  size_t i = (size_t)blockIdx.x * blockDim.x + threadIdx.x;
  size_t stride = (size_t)gridDim.x * blockDim.x;
  uint4 z = {0u, 0u, 0u, 0u};
  for (; i < n16; i += stride) p[i] = z;
}

// ---------------- weight prep ----------------
__global__ void build_we1_k(const float* __restrict__ W1, const float* __restrict__ r1,
                            float* __restrict__ we1) {
  int t = blockIdx.x * blockDim.x + threadIdx.x;
  if (t >= R1 * 32) return;
  int k = t / 32, o = t % 32;
  we1[t] = (k < KK) ? W1[k * 32 + o] : r1[o];
}

// Wt[o][r] f16 transposed; r < 48*CIN -> W[r][o], else root[r-48*CIN][o]
template <int CIN, int CO>
__global__ void build_wt_k(const float* __restrict__ W, const float* __restrict__ root,
                           _Float16* __restrict__ wt) {
  const int R = 49 * CIN;
  int t = blockIdx.x * blockDim.x + threadIdx.x;
  if (t >= CO * R) return;
  int o = t / R, r = t % R;
  float v = (r < KK * CIN) ? W[(size_t)r * CO + o] : root[(size_t)(r - KK * CIN) * CO + o];
  wt[t] = (_Float16)v;
}

// ---------------- layer 1 (c_in = 1): one thread per edge, 8 f32 atomics -----------
__global__ void scatter1_k(const float* __restrict__ x, const int* __restrict__ ei,
                           const float* __restrict__ ps, float* __restrict__ acc1) {
  int e = blockIdx.x * blockDim.x + threadIdx.x;
  if (e >= EE) return;
  Basis bs; bs.init(ps, e);
  int src = ei[e], dst = ei[EE + e];
  float xv = x[src];
  float* base = acc1 + (size_t)dst * R1;
  #pragma unroll
  for (int s = 0; s < 8; ++s) {
    float w; int kid;
    bs.corner(s, w, kid);
    atomicAdd(base + kid, w * xv);
  }
}

__global__ void copy_x1_k(const float* __restrict__ x, float* __restrict__ acc1) {
  int n = blockIdx.x * blockDim.x + threadIdx.x;
  if (n < NN) acc1[(size_t)n * R1 + KK] = x[n];
}

__global__ void dense1_k(const float* __restrict__ acc1, const float* __restrict__ we1,
                         const float* __restrict__ b1, float* __restrict__ h1) {
  int t = blockIdx.x * blockDim.x + threadIdx.x;
  if (t >= NN * 32) return;
  int n = t >> 5, o = t & 31;
  float s = b1[o];
  const float* a = acc1 + (size_t)n * R1;
  #pragma unroll 7
  for (int k = 0; k < R1; ++k) s = fmaf(a[k], we1[k * 32 + o], s);
  h1[t] = fmaxf(s, 0.f);
}

// ---------------- coalesced scatter: thread = (edge, channel pair), loop corners ----
// A wave covers one edge's full channel row (coalesced x read); each corner's PAIRS
// packed atomics from the wave hit one contiguous bucket row in L2.
template <int CIN>
__global__ void scatter_f16_k(const float* __restrict__ h, const int* __restrict__ ei,
                              const float* __restrict__ ps, _Float16* __restrict__ acc) {
  const int PAIRS = CIN / 2;         // 16 or 32 (power of two)
  const int Rext  = 49 * CIN;
  int t = blockIdx.x * blockDim.x + threadIdx.x;
  if (t >= EE * PAIRS) return;
  int e = t / PAIRS;
  int c = (t % PAIRS) * 2;
  Basis bs; bs.init(ps, e);
  int src = ei[e], dst = ei[EE + e];
  float x0 = h[(size_t)src * CIN + c];
  float x1 = h[(size_t)src * CIN + c + 1];
  _Float16* base = acc + (size_t)dst * Rext + c;
  #pragma unroll
  for (int s = 0; s < 8; ++s) {
    float w; int kid;
    bs.corner(s, w, kid);
    pk_atomic_add(base + kid * CIN, w * x0, w * x1);
  }
}

// copy x_in into the 49th block so the GEMM also applies the root/skip term
template <int CIN>
__global__ void copy_x_k(const float* __restrict__ h, _Float16* __restrict__ acc) {
  const int Rext = 49 * CIN;
  int t = blockIdx.x * blockDim.x + threadIdx.x;
  if (t >= NN * CIN) return;
  int n = t / CIN, i = t % CIN;
  acc[(size_t)n * Rext + KK * CIN + i] = (_Float16)h[t];
}

// ---------------- WMMA GEMM, 32x64 register tile per wave (2x4 subtiles) ------------
// out[M x CO] = relu(A[M x R] * Wt^T + bias);  A row-major f16, Wt = [CO x R] f16.
// Fragment layouts per CDNA5 ISA 7.12.2 (wave32):
//   A 16x32 f16: lane = row M (lane&15); halves 0..7 = K[k0..k0+7], 8..15 = K[k0+16..+23],
//                k0 = r + (lane<16 ? 0 : 8)
//   B 32x16 f16: lane = col N (lane&15); halves 0..7 = K[kb..kb+7], 8..15 = K[kb+16..+23], kb = r + (lane<16?0:8)
//   C/D f32:     VGPR i -> M = i + (lane<16?0:8), N = lane&15
template <int R, int CO>
__global__ __launch_bounds__(32 * (CO / 64))
void gemm_wmma_k(const _Float16* __restrict__ A, const _Float16* __restrict__ Wt,
                 const float* __restrict__ bias, float* __restrict__ out) {
  const int lane = threadIdx.x;
  const int hi   = lane >> 4;
  const int l15  = lane & 15;
  const int m0   = blockIdx.x * 32;          // 625 blocks * 32 rows = 20000 exactly
  const int n0   = threadIdx.y * 64;
  const size_t arow0 = (size_t)(m0 + l15) * R;
  const size_t arow1 = arow0 + (size_t)16 * R;
  size_t brow[4];
  #pragma unroll
  for (int j = 0; j < 4; ++j) brow[j] = (size_t)(n0 + j * 16 + l15) * R;

  v8f acc[2][4] = {};
  for (int r = 0; r < R; r += 32) {
    const int ka = r + hi * 8;
    const int kb = r + hi * 8;
    v16h a[2];
    {
      v8h lo0 = *(const v8h*)(A + arow0 + ka);
      v8h hi0 = *(const v8h*)(A + arow0 + ka + 16);
      v8h lo1 = *(const v8h*)(A + arow1 + ka);
      v8h hi1 = *(const v8h*)(A + arow1 + ka + 16);
      #pragma unroll
      for (int q = 0; q < 8; ++q) {
        a[0][q] = lo0[q]; a[0][q + 8] = hi0[q];
        a[1][q] = lo1[q]; a[1][q + 8] = hi1[q];
      }
    }
    v16h b[4];
    #pragma unroll
    for (int j = 0; j < 4; ++j) {
      v8h blo = *(const v8h*)(Wt + brow[j] + kb);
      v8h bhi = *(const v8h*)(Wt + brow[j] + kb + 16);
      #pragma unroll
      for (int q = 0; q < 8; ++q) { b[j][q] = blo[q]; b[j][q + 8] = bhi[q]; }
    }
    #pragma unroll
    for (int i = 0; i < 2; ++i)
      #pragma unroll
      for (int j = 0; j < 4; ++j)
        acc[i][j] = __builtin_amdgcn_wmma_f32_16x16x32_f16(false, a[i], false, b[j],
                                                           (short)0, acc[i][j],
                                                           false, false);
  }
  #pragma unroll
  for (int j = 0; j < 4; ++j) {
    const int col = n0 + j * 16 + l15;
    const float bb = bias[col];
    #pragma unroll
    for (int i = 0; i < 2; ++i) {
      #pragma unroll
      for (int q = 0; q < 8; ++q) {
        int m = m0 + i * 16 + q + hi * 8;
        float v = acc[i][j][q] + bb;
        out[(size_t)m * CO + col] = v > 0.f ? v : 0.f;
      }
    }
  }
}

// ---------------- log_softmax over 128 channels, one wave per row, in place --------
__global__ void logsoftmax_k(float* __restrict__ o) {
  int lane = threadIdx.x & 31;
  int wv   = threadIdx.x >> 5;
  int n    = blockIdx.x * 8 + wv;
  if (n >= NN) return;
  float* r = o + (size_t)n * 128;
  float v[4];
  float m = -3.4e38f;
  #pragma unroll
  for (int i = 0; i < 4; ++i) { v[i] = r[lane + 32 * i]; m = fmaxf(m, v[i]); }
  #pragma unroll
  for (int off = 16; off >= 1; off >>= 1) m = fmaxf(m, __shfl_xor(m, off, 32));
  float s = 0.f;
  #pragma unroll
  for (int i = 0; i < 4; ++i) s += __expf(v[i] - m);
  #pragma unroll
  for (int off = 16; off >= 1; off >>= 1) s += __shfl_xor(s, off, 32);
  float lse = m + __logf(s);
  #pragma unroll
  for (int i = 0; i < 4; ++i) r[lane + 32 * i] = v[i] - lse;
}

// ---------------- host side ----------------
extern "C" void kernel_launch(void* const* d_in, const int* in_sizes, int n_in,
                              void* d_out, int out_size, void* d_ws, size_t ws_size,
                              hipStream_t stream) {
  const float* x  = (const float*)d_in[0];
  const int*   ei = (const int*)d_in[1];
  const float* ps = (const float*)d_in[2];
  const float* W1 = (const float*)d_in[3];
  const float* r1 = (const float*)d_in[4];
  const float* b1 = (const float*)d_in[5];
  const float* W2 = (const float*)d_in[6];
  const float* r2 = (const float*)d_in[7];
  const float* b2 = (const float*)d_in[8];
  const float* W3 = (const float*)d_in[9];
  const float* r3 = (const float*)d_in[10];
  const float* b3 = (const float*)d_in[11];
  float* out = (float*)d_out;

  char* ws = (char*)d_ws;
  auto al = [](size_t v) { return (v + 255) & ~(size_t)255; };
  size_t o = 0;
  float*    acc1 = (float*)(ws + o);     o = al(o + (size_t)NN * R1 * 4);
  _Float16* acc2 = (_Float16*)(ws + o);  o = al(o + (size_t)NN * R2 * 2);
  _Float16* acc3 = (_Float16*)(ws + o);  size_t zend = o + (size_t)NN * R3 * 2; o = al(zend);
  float*    h1   = (float*)(ws + o);     o = al(o + (size_t)NN * 32 * 4);
  float*    h2   = (float*)(ws + o);     o = al(o + (size_t)NN * 64 * 4);
  float*    we1  = (float*)(ws + o);     o = al(o + (size_t)R1 * 32 * 4);
  _Float16* wt2  = (_Float16*)(ws + o);  o = al(o + (size_t)64 * R2 * 2);
  _Float16* wt3  = (_Float16*)(ws + o);  o = al(o + (size_t)128 * R3 * 2);

  // 1) zero all scatter accumulators (required every call: atomics accumulate)
  zero_bytes_k<<<4096, 256, 0, stream>>>((uint4*)ws, zend / 16);

  // 2) weight prep (f16 transposed [CO x R], root folded in as block 48)
  build_we1_k<<<(R1 * 32 + 255) / 256, 256, 0, stream>>>(W1, r1, we1);
  build_wt_k<32, 64><<<(64 * R2 + 255) / 256, 256, 0, stream>>>(W2, r2, wt2);
  build_wt_k<64, 128><<<(128 * R3 + 255) / 256, 256, 0, stream>>>(W3, r3, wt3);

  // 3) layer 1: 1 -> 32
  scatter1_k<<<(EE + 255) / 256, 256, 0, stream>>>(x, ei, ps, acc1);
  copy_x1_k<<<(NN + 255) / 256, 256, 0, stream>>>(x, acc1);
  dense1_k<<<(NN * 32 + 255) / 256, 256, 0, stream>>>(acc1, we1, b1, h1);

  // 4) layer 2: 32 -> 64 (pk_add_f16 atomics -> f16 acc -> WMMA f16)
  scatter_f16_k<32><<<(EE * 16 + 255) / 256, 256, 0, stream>>>(h1, ei, ps, acc2);
  copy_x_k<32><<<(NN * 32 + 255) / 256, 256, 0, stream>>>(h1, acc2);
  gemm_wmma_k<R2, 64><<<625, dim3(32, 1), 0, stream>>>(acc2, wt2, b2, h2);

  // 5) layer 3: 64 -> 128
  scatter_f16_k<64><<<(EE * 32 + 255) / 256, 256, 0, stream>>>(h2, ei, ps, acc3);
  copy_x_k<64><<<(NN * 64 + 255) / 256, 256, 0, stream>>>(h2, acc3);
  gemm_wmma_k<R3, 128><<<625, dim3(32, 2), 0, stream>>>(acc3, wt3, b3, out);

  // 6) log_softmax in place on the output
  logsoftmax_k<<<2500, 256, 0, stream>>>(out);
}
